// LinearBlock_42262478193307
// MI455X (gfx1250) — matrix operand order repaired and checked by reading, not compiled, not run on hardware
//
#include <hip/hip_runtime.h>

// ---------------------------------------------------------------------------
// Problem constants (from reference): B=4, T=8192, D=768, H=12, HD=64, DFF=3072
// ---------------------------------------------------------------------------
#define BB   4
#define TT   8192
#define DD   768
#define HH   12
#define HD   64
#define DFF  3072
#define MM   (BB * TT)          // 32768 rows
#define BH   (BB * HH)          // 48

typedef unsigned short u16;
typedef __attribute__((ext_vector_type(16))) __bf16 bf16x16;
typedef __attribute__((ext_vector_type(8)))  float  floatx8;
typedef int v4i __attribute__((vector_size(16)));   // matches builtin param type

// ---- CDNA5 async global->LDS path (guarded; falls back to sync copy) -------
#if defined(__HIP_DEVICE_COMPILE__) && __has_builtin(__builtin_amdgcn_global_load_async_to_lds_b128)
#define HAS_ASYNC_LDS 1
#else
#define HAS_ASYNC_LDS 0
#endif

__device__ __forceinline__ void copy16_to_lds(const u16* g, u16* l) {
#if HAS_ASYNC_LDS
    __builtin_amdgcn_global_load_async_to_lds_b128(
        (__attribute__((address_space(1))) v4i*)(size_t)g,
        (__attribute__((address_space(3))) v4i*)(unsigned)(size_t)l,
        0, 0);
#else
    *(uint4*)l = *(const uint4*)g;
#endif
}
__device__ __forceinline__ void async_wait_keep_prefetch() { // <=4 outstanding
#if HAS_ASYNC_LDS
# if __has_builtin(__builtin_amdgcn_s_wait_asynccnt)
    __builtin_amdgcn_s_wait_asynccnt(4);
# else
    asm volatile("s_wait_asynccnt 4" ::: "memory");
# endif
#endif
}
__device__ __forceinline__ void async_wait_all() {
#if HAS_ASYNC_LDS
# if __has_builtin(__builtin_amdgcn_s_wait_asynccnt)
    __builtin_amdgcn_s_wait_asynccnt(0);
# else
    asm volatile("s_wait_asynccnt 0" ::: "memory");
# endif
#endif
}

// ---------------- bf16 helpers (storage = unsigned short) -------------------
__device__ __forceinline__ u16 f2bf(float f) {
    union { float f; unsigned u; } x; x.f = f;
    unsigned r = x.u + 0x7fffu + ((x.u >> 16) & 1u);   // round-to-nearest-even
    return (u16)(r >> 16);
}
__device__ __forceinline__ float bf2f(u16 h) {
    union { unsigned u; float f; } x; x.u = ((unsigned)h) << 16;
    return x.f;
}
__device__ __forceinline__ float elu1(float x) {      // elu(x)+1
    return x > 0.f ? x + 1.f : __expf(x);
}

// ------------------- WMMA bf16 fragment loaders -----------------------------
// A fragment (16x32, M x K), contiguous k order used by this code (not the ISA table order):
//   lane L: row M = L&15 ; group g = L>>4
//   v0..v7 hold K = 16g..16g+15 (contiguous, matching B)
union Frag { uint4 q[2]; bf16x16 v; };

__device__ __forceinline__ bf16x16 load_a_frag(const u16* A, int lda, int row,
                                               int k, int lane) {
    int m  = row + (lane & 15);
    int kb = k + ((lane >> 4) << 4);
    const u16* p = A + (size_t)m * lda + kb;
    Frag f;
    f.q[0] = *(const uint4*)(p);
    f.q[1] = *(const uint4*)(p + 8);
    return f.v;
}
// B fragment (32x16, K x N) from a pre-transposed Bt[N][K]:
//   lane L: col N = L&15 ; lanes 0-15 hold K = k..k+15, lanes 16-31 K = k+16..k+31
__device__ __forceinline__ bf16x16 load_b_frag(const u16* Bt, int ldb, int n,
                                               int k, int lane) {
    int nn = n + (lane & 15);
    int kb = k + ((lane >> 4) << 4);
    const uint4* p = (const uint4*)(Bt + (size_t)nn * ldb + kb);
    Frag f; f.q[0] = p[0]; f.q[1] = p[1];
    return f.v;
}
// LDS-staged fragment loaders; K-slice tiles are [row][32+8pad] (pad kills the
// 4-way ds bank conflict: stride 80B -> 16 distinct banks across lanes 0-15).
#define KPAD 40
__device__ __forceinline__ bf16x16 lds_a_frag(const u16* tile, int row, int lane) {
    int r  = row + (lane & 15);
    int kb = (lane >> 4) << 4;
    const u16* p = tile + r * KPAD + kb;
    Frag f; f.q[0] = *(const uint4*)p; f.q[1] = *(const uint4*)(p + 8);
    return f.v;
}
__device__ __forceinline__ bf16x16 lds_b_frag(const u16* tile, int n, int lane) {
    int nn = n + (lane & 15);
    int kb = (lane >> 4) << 4;
    const u16* p = tile + nn * KPAD + kb;
    Frag f; f.q[0] = *(const uint4*)p; f.q[1] = *(const uint4*)(p + 8);
    return f.v;
}

// ---------------------------------------------------------------------------
// Generic WMMA GEMM:  C[M x N] = A[M x K] * Bt[N x K]^T  (+ fused epilogue)
// block = 256 threads = 8 waves (4 along M x 2 along N).
// block tile = 128(M) x 128(N); wave tile = 32(M) x 64(N) (8 accumulators).
// K staged through double-buffered LDS via async global->LDS copies.
// grid.x = M/128, grid.y = N/128.
// ---------------------------------------------------------------------------
enum { EPI_F32_BIAS_RES = 0, EPI_Q = 1, EPI_KT = 2, EPI_VT = 3, EPI_GELU = 4 };

template <int EPI>
__global__ __launch_bounds__(256) void gemm_bf16(
    const u16* __restrict__ A, const u16* __restrict__ Bt,
    int K, int lda, int ldb, int N,
    const float* __restrict__ bias, const float* __restrict__ resid,
    float* __restrict__ outF, u16* __restrict__ outB)
{
    __shared__ u16 Asm[2][128 * KPAD];
    __shared__ u16 Bsm[2][128 * KPAD];

    const int tid  = threadIdx.x;
    const int lane = tid & 31;
    const int w    = tid >> 5;
    const int m0   = blockIdx.x * 128;
    const int n0   = blockIdx.y * 128;
    const int wm   = (w & 3) * 32;      // wave row offset in block tile
    const int wn   = (w >> 2) * 64;     // wave col offset in block tile

    // per-thread copy chunks: 512 x 16B per tile, 2 A + 2 B chunks per thread
    const int r0  = tid >> 2,          kk0 = (tid & 3) << 3;
    const int r1  = (tid + 256) >> 2,  kk1 = (tid & 3) << 3;

    floatx8 acc[2][4] = {};

    auto issue = [&](int k, int nb) {
        copy16_to_lds(A  + (size_t)(m0 + r0) * lda + k + kk0, &Asm[nb][r0 * KPAD + kk0]);
        copy16_to_lds(A  + (size_t)(m0 + r1) * lda + k + kk1, &Asm[nb][r1 * KPAD + kk1]);
        copy16_to_lds(Bt + (size_t)(n0 + r0) * ldb + k + kk0, &Bsm[nb][r0 * KPAD + kk0]);
        copy16_to_lds(Bt + (size_t)(n0 + r1) * ldb + k + kk1, &Bsm[nb][r1 * KPAD + kk1]);
    };

    issue(0, 0);
    const int nsteps = K / 32;
    for (int s = 0; s < nsteps; ++s) {
        const int buf = s & 1;
        if (s + 1 < nsteps) {           // prefetch next K-slice, then wait cur
            issue((s + 1) * 32, buf ^ 1);
            async_wait_keep_prefetch();
        } else {
            async_wait_all();
        }
        __syncthreads();                // all waves' stage-s copies visible

        const u16* As = Asm[buf];
        const u16* Bs = Bsm[buf];
        bf16x16 a0 = lds_a_frag(As, wm,      lane);
        bf16x16 a1 = lds_a_frag(As, wm + 16, lane);
#pragma unroll
        for (int j = 0; j < 4; ++j) {
            bf16x16 b = lds_b_frag(Bs, wn + j * 16, lane);
            acc[0][j] = __builtin_amdgcn_wmma_f32_16x16x32_bf16(
                false, a0, false, b, (short)0, acc[0][j], false, false);
            acc[1][j] = __builtin_amdgcn_wmma_f32_16x16x32_bf16(
                false, a1, false, b, (short)0, acc[1][j], false, false);
        }
        __syncthreads();                // safe to overwrite buf next iteration
    }

    // C/D layout: col = n + (lane&15); VGPR i -> row = m + i + 8*(lane>>4)
    const int colLane = lane & 15;
    const int rOff    = (lane >> 4) * 8;

#pragma unroll
    for (int sub = 0; sub < 2; ++sub) {
        const int m = m0 + wm + sub * 16;
#pragma unroll
        for (int j = 0; j < 4; ++j) {
            const int c = n0 + wn + j * 16 + colLane;
            if (EPI == EPI_F32_BIAS_RES) {
                const float bv = bias[c];
#pragma unroll
                for (int i = 0; i < 8; ++i) {
                    const int r = m + rOff + i;
                    outF[(size_t)r * N + c] = acc[sub][j][i] + bv + resid[(size_t)r * N + c];
                }
            } else if (EPI == EPI_GELU) {
                const float bv = bias[c];
#pragma unroll
                for (int i = 0; i < 8; ++i) {
                    const int r = m + rOff + i;
                    float v = acc[sub][j][i] + bv;
                    float g = 0.5f * v * (1.f + erff(v * 0.70710678118654752f));
                    outB[(size_t)r * N + c] = f2bf(g);
                }
            } else if (EPI == EPI_Q) {
                const int h = c >> 6, d = c & 63;
#pragma unroll
                for (int i = 0; i < 8; ++i) {
                    const int r = m + rOff + i;
                    const int b_ = r >> 13, t = r & (TT - 1);
                    outB[(((size_t)b_ * HH + h) * TT + t) * HD + d] = f2bf(elu1(acc[sub][j][i]));
                }
            } else { // EPI_KT / EPI_VT : transposed bf16 store -> [bh][d][T]
                const int h = c >> 6, d = c & 63;
                const int rr0 = m + rOff;
                const int b_ = rr0 >> 13, t0 = rr0 & (TT - 1);
                union { u16 s[8]; uint4 q[1]; } pack;
#pragma unroll
                for (int i = 0; i < 8; ++i) {
                    float v = acc[sub][j][i];
                    if (EPI == EPI_KT) v = elu1(v);
                    pack.s[i] = f2bf(v);
                }
                u16* dst = outB + (((size_t)b_ * HH + h) * HD + d) * TT + t0;
                *(uint4*)dst = pack.q[0];
            }
        }
    }
}

// ---------------------------------------------------------------------------
// KV split-K GEMM:  KV[bh][d][e] += sum_t K^T[d,t] * V[t,e]   (f32 atomics)
// grid = (BH, 16 K-slices), 8 waves: 4(M=d) x 2(N=e halves), each wave 16x32.
// ---------------------------------------------------------------------------
__global__ __launch_bounds__(256) void kv_gemm(
    const u16* __restrict__ Kt, const u16* __restrict__ Vt,
    float* __restrict__ KVf)
{
    const int lane = threadIdx.x & 31;
    const int w    = threadIdx.x >> 5;
    const int bh   = blockIdx.x;
    const int md   = (w & 3) * 16;
    const int ne0  = (w >> 2) * 32;
    const u16* Ab  = Kt + (size_t)bh * HD * TT;
    const u16* Bb  = Vt + (size_t)bh * HD * TT;

    floatx8 acc[2] = {};
    const int t0 = blockIdx.y * (TT / 16);
#pragma unroll 4
    for (int k = t0; k < t0 + TT / 16; k += 32) {
        bf16x16 a = load_a_frag(Ab, TT, md, k, lane);
#pragma unroll
        for (int j = 0; j < 2; ++j) {
            bf16x16 b = load_b_frag(Bb, TT, ne0 + j * 16, k, lane);
            acc[j] = __builtin_amdgcn_wmma_f32_16x16x32_bf16(
                false, a, false, b, (short)0, acc[j], false, false);
        }
    }
    const int colLane = lane & 15, rOff = (lane >> 4) * 8;
#pragma unroll
    for (int j = 0; j < 2; ++j) {
        const int e = ne0 + j * 16 + colLane;
#pragma unroll
        for (int i = 0; i < 8; ++i) {
            const int d = md + rOff + i;
            atomicAdd(&KVf[(size_t)bh * 4096 + d * 64 + e], acc[j][i]);
        }
    }
}

// ---------------------------------------------------------------------------
// ctx GEMM: ctx[b,t,h*64+e] = (Q[bh,t,:] @ KV[bh]) * Zinv[bh,t]
// grid = (T/128, BH); wave = 16(t) x 64(e), K = 64 (2 steps).
// ---------------------------------------------------------------------------
__global__ __launch_bounds__(256) void ctx_gemm(
    const u16* __restrict__ Q, const u16* __restrict__ KVt,
    const float* __restrict__ Zinv, u16* __restrict__ ctx)
{
    const int lane = threadIdx.x & 31;
    const int w    = threadIdx.x >> 5;
    const int bh   = blockIdx.y;
    const int b_   = bh / HH, h = bh % HH;
    const int t0   = blockIdx.x * 128 + w * 16;
    const u16* Ab  = Q + (size_t)bh * TT * HD;
    const u16* Bb  = KVt + (size_t)bh * 4096;   // KV^T : [e][d]

    floatx8 acc[4] = {};
#pragma unroll
    for (int k = 0; k < HD; k += 32) {
        bf16x16 a = load_a_frag(Ab, HD, t0, k, lane);
#pragma unroll
        for (int j = 0; j < 4; ++j) {
            bf16x16 b = load_b_frag(Bb, HD, j * 16, k, lane);
            acc[j] = __builtin_amdgcn_wmma_f32_16x16x32_bf16(
                false, a, false, b, (short)0, acc[j], false, false);
        }
    }
    const int colLane = lane & 15, rOff = (lane >> 4) * 8;
#pragma unroll
    for (int j = 0; j < 4; ++j) {
        const int e = j * 16 + colLane;
#pragma unroll
        for (int i = 0; i < 8; ++i) {
            const int t = t0 + rOff + i;
            const float zi = Zinv[(size_t)bh * TT + t];
            ctx[((size_t)b_ * TT + t) * DD + h * HD + e] = f2bf(acc[j][i] * zi);
        }
    }
}

// ----------------------------- small kernels --------------------------------
// LayerNorm over D=768; one wave per row, 4 rows per 128-thread block.
__global__ __launch_bounds__(128) void ln_kernel(
    const float* __restrict__ x, const float* __restrict__ g,
    const float* __restrict__ bta, u16* __restrict__ out)
{
    const int lane = threadIdx.x & 31;
    const int w    = threadIdx.x >> 5;
    const int row  = blockIdx.x * 4 + w;
    const float* xr = x + (size_t)row * DD;
    float s = 0.f, ss = 0.f;
    for (int j = lane; j < DD; j += 32) { float v = xr[j]; s += v; ss += v * v; }
#pragma unroll
    for (int o = 16; o > 0; o >>= 1) {
        s  += __shfl_xor(s, o, 32);
        ss += __shfl_xor(ss, o, 32);
    }
    const float mu  = s * (1.f / DD);
    const float var = ss * (1.f / DD) - mu * mu;
    const float rs  = rsqrtf(var + 1e-5f);
    u16* orow = out + (size_t)row * DD;
    for (int j = lane; j < DD; j += 32)
        orow[j] = f2bf((xr[j] - mu) * rs * g[j] + bta[j]);
}

// K_sum[bh*64+d] = sum_t Kt[bh,d,t]; one wave per row.
__global__ __launch_bounds__(256) void ksum_kernel(
    const u16* __restrict__ Kt, float* __restrict__ Ksum)
{
    const int lane = threadIdx.x & 31;
    const int w    = threadIdx.x >> 5;
    const int row  = blockIdx.x * 8 + w;             // 0..BH*64-1
    const u16* p = Kt + (size_t)row * TT;
    float s = 0.f;
    for (int t = lane; t < TT; t += 32) s += bf2f(p[t]);
#pragma unroll
    for (int o = 16; o > 0; o >>= 1) s += __shfl_xor(s, o, 32);
    if (lane == 0) Ksum[row] = s;
}

// Zinv[bh,t] = 1 / (eps + Q[bh,t,:] . Ksum[bh,:])
__global__ __launch_bounds__(256) void zinv_kernel(
    const u16* __restrict__ Q, const float* __restrict__ Ksum,
    float* __restrict__ Zinv)
{
    const int idx = blockIdx.x * 256 + threadIdx.x;  // bh*T + t
    const int bh  = idx >> 13;
    const u16* q  = Q + (size_t)idx * HD;
    const float* ks = Ksum + bh * HD;
    float s = 1e-6f;
#pragma unroll 8
    for (int d = 0; d < HD; ++d) s += bf2f(q[d]) * ks[d];
    Zinv[idx] = 1.f / s;
}

__global__ void zero_kernel(float* p, int n) {
    int i = blockIdx.x * 256 + threadIdx.x;
    if (i < n) p[i] = 0.f;
}

// KVt[bh][e][d] = bf16(KVf[bh][d][e])
__global__ void kvconv_kernel(const float* __restrict__ KVf, u16* __restrict__ KVt) {
    int idx = blockIdx.x * 256 + threadIdx.x;        // BH*4096
    int bh = idx >> 12, r = idx & 4095;
    int e = r >> 6, d = r & 63;
    KVt[idx] = f2bf(KVf[(size_t)bh * 4096 + d * 64 + e]);
}

// Wt[n*K + k] = bf16(W[k*N + n])
__global__ void wtrans_kernel(const float* __restrict__ W, u16* __restrict__ Wt,
                              int Kdim, int Ndim) {
    int idx = blockIdx.x * 256 + threadIdx.x;
    if (idx >= Kdim * Ndim) return;
    int n = idx / Kdim, k = idx - n * Kdim;
    Wt[idx] = f2bf(W[(size_t)k * Ndim + n]);
}

// ---------------------------------------------------------------------------
extern "C" void kernel_launch(void* const* d_in, const int* in_sizes, int n_in,
                              void* d_out, int out_size, void* d_ws, size_t ws_size,
                              hipStream_t stream) {
    const float* x     = (const float*)d_in[0];
    const float* Wq    = (const float*)d_in[1];
    const float* Wk    = (const float*)d_in[2];
    const float* Wv    = (const float*)d_in[3];
    const float* Wo    = (const float*)d_in[4];
    const float* bo    = (const float*)d_in[5];
    const float* ln1_g = (const float*)d_in[6];
    const float* ln1_b = (const float*)d_in[7];
    const float* ln2_g = (const float*)d_in[8];
    const float* ln2_b = (const float*)d_in[9];
    const float* W1    = (const float*)d_in[10];
    const float* b1    = (const float*)d_in[11];
    const float* W2    = (const float*)d_in[12];
    const float* b2    = (const float*)d_in[13];
    float* out = (float*)d_out;

    // ---- workspace carve-out ----
    char* base = (char*)d_ws;
    size_t off = 0;
    auto alloc = [&](size_t bytes) -> void* {
        void* p = base + off;
        off += (bytes + 255) & ~(size_t)255;
        return p;
    };
    u16*   xn   = (u16*)  alloc((size_t)MM * DD * 2);   // LN1 out, reused as LN2 out
    u16*   WqT  = (u16*)  alloc((size_t)DD * DD * 2);
    u16*   WkT  = (u16*)  alloc((size_t)DD * DD * 2);
    u16*   WvT  = (u16*)  alloc((size_t)DD * DD * 2);
    u16*   WoT  = (u16*)  alloc((size_t)DD * DD * 2);
    u16*   W1T  = (u16*)  alloc((size_t)DD * DFF * 2);  // [DFF][DD]
    u16*   W2T  = (u16*)  alloc((size_t)DD * DFF * 2);  // [DD][DFF]
    u16*   Qbf  = (u16*)  alloc((size_t)MM * DD * 2);   // [bh][t][d]
    u16*   Kt   = (u16*)  alloc((size_t)MM * DD * 2);   // [bh][d][t], reused as ctx
    u16*   Vt   = (u16*)  alloc((size_t)MM * DD * 2);   // [bh][e][t]
    float* KVf  = (float*)alloc((size_t)BH * 4096 * 4);
    u16*   KVt  = (u16*)  alloc((size_t)BH * 4096 * 2);
    float* Ksum = (float*)alloc((size_t)BH * HD * 4);
    float* Zinv = (float*)alloc((size_t)BH * TT * 4);
    float* y1   = (float*)alloc((size_t)MM * DD * 4);
    u16*   h1   = (u16*)  alloc((size_t)MM * DFF * 2);
    u16*   ctx  = Kt;                                   // reuse
    u16*   y1n  = xn;                                   // reuse
    (void)ws_size; (void)in_sizes; (void)n_in; (void)out_size;

    // ---- 1. weight convert + transpose to bf16 [N][K] ----
    wtrans_kernel<<<(DD * DD) / 256, 256, 0, stream>>>(Wq, WqT, DD, DD);
    wtrans_kernel<<<(DD * DD) / 256, 256, 0, stream>>>(Wk, WkT, DD, DD);
    wtrans_kernel<<<(DD * DD) / 256, 256, 0, stream>>>(Wv, WvT, DD, DD);
    wtrans_kernel<<<(DD * DD) / 256, 256, 0, stream>>>(Wo, WoT, DD, DD);
    wtrans_kernel<<<(DD * DFF) / 256, 256, 0, stream>>>(W1, W1T, DD, DFF);   // Wt[DFF][DD]
    wtrans_kernel<<<(DD * DFF) / 256, 256, 0, stream>>>(W2, W2T, DFF, DD);   // Wt[DD][DFF]

    // ---- 2. LN1 ----
    ln_kernel<<<MM / 4, 128, 0, stream>>>(x, ln1_g, ln1_b, xn);

    // ---- 3-5. Q/K/V projections (fused elu+1 and layout transforms) ----
    dim3 gD(MM / 128, DD / 128);
    gemm_bf16<EPI_Q ><<<gD, 256, 0, stream>>>(xn, WqT, DD, DD, DD, DD, nullptr, nullptr, nullptr, Qbf);
    gemm_bf16<EPI_KT><<<gD, 256, 0, stream>>>(xn, WkT, DD, DD, DD, DD, nullptr, nullptr, nullptr, Kt);
    gemm_bf16<EPI_VT><<<gD, 256, 0, stream>>>(xn, WvT, DD, DD, DD, DD, nullptr, nullptr, nullptr, Vt);

    // ---- 6-9. linear attention state: KV = K^T V (split-K atomics), K_sum ----
    zero_kernel<<<(BH * 4096 + 255) / 256, 256, 0, stream>>>(KVf, BH * 4096);
    kv_gemm<<<dim3(BH, 16), 256, 0, stream>>>(Kt, Vt, KVf);
    ksum_kernel<<<(BH * HD) / 8, 256, 0, stream>>>(Kt, Ksum);
    kvconv_kernel<<<(BH * 4096) / 256, 256, 0, stream>>>(KVf, KVt);

    // ---- 10. Z = Q . K_sum + eps ----
    zinv_kernel<<<(BH * TT) / 256, 256, 0, stream>>>(Qbf, Ksum, Zinv);

    // ---- 11. ctx = (Q @ KV) / Z  -> [B*T, D] bf16 ----
    ctx_gemm<<<dim3(TT / 128, BH), 256, 0, stream>>>(Qbf, KVt, Zinv, ctx);

    // ---- 12. att = ctx @ Wo + bo + x  (f32) ----
    gemm_bf16<EPI_F32_BIAS_RES><<<gD, 256, 0, stream>>>(ctx, WoT, DD, DD, DD, DD, bo, x, y1, nullptr);

    // ---- 13. LN2 ----
    ln_kernel<<<MM / 4, 128, 0, stream>>>(y1, ln2_g, ln2_b, y1n);

    // ---- 14. h1 = gelu(y1n @ W1 + b1)  (bf16) ----
    gemm_bf16<EPI_GELU><<<dim3(MM / 128, DFF / 128), 256, 0, stream>>>(
        y1n, W1T, DD, DD, DD, DFF, b1, nullptr, nullptr, h1);

    // ---- 15. out = h1 @ W2 + b2 + y1  (f32) ----
    gemm_bf16<EPI_F32_BIAS_RES><<<gD, 256, 0, stream>>>(
        h1, W2T, DFF, DFF, DFF, DD, b2, y1, out, nullptr);
}
